// CADense_73426760892514
// MI455X (gfx1250) — matrix operand order repaired and checked by reading, not compiled, not run on hardware
//
#include <hip/hip_runtime.h>

// ---------------------------------------------------------------------------
// CADense: out = relu( ((data@u) * (s + context@w)) @ v^T + 2*bias )
// B=8192, N_IN=2048, UNITS=2048, RANK=256, C_CTX=512
// Split-bf16 (hi/lo) GEMMs on v_wmma_f32_16x16x32_bf16, f32 accumulate.
// A: register double-buffered staging with on-the-fly f32->hi/lo bf16 split.
// B: async global->LDS copies (ASYNCcnt) into ping-pong LDS buffers when the
//    toolchain exposes the gfx1250 async builtins; register staging otherwise.
// ---------------------------------------------------------------------------

#if defined(__has_builtin)
#  if __has_builtin(__builtin_amdgcn_global_load_async_to_lds_b128)
#    define ASYNC_LDS 1
#  endif
#endif
#ifndef ASYNC_LDS
#  define ASYNC_LDS 0
#endif

typedef __attribute__((ext_vector_type(16))) __bf16 bf16x16;
typedef __attribute__((ext_vector_type(8)))  __bf16 bf16x8;
typedef __attribute__((ext_vector_type(4)))  __bf16 bf16x4;
typedef __attribute__((ext_vector_type(8)))  float  f32x8;
typedef __attribute__((ext_vector_type(4)))  float  f32x4;
typedef __attribute__((ext_vector_type(4)))  unsigned u32x4;

#if ASYNC_LDS
// async-to-LDS builtins take vector-typed, address-space-qualified pointers:
//   b128: (v4i addrspace(1)*, v4i addrspace(3)*, imm offset, imm cpol)
typedef __attribute__((ext_vector_type(4))) int v4i;
typedef __attribute__((address_space(1))) v4i* as1_v4i;
typedef __attribute__((address_space(3))) v4i* as3_v4i;
#endif

#define BATCH   8192
#define NIN     2048
#define UNITS_N 2048
#define RANK_N  256
#define CCTX    512

// ---------------- prep: f32 -> (hi, lo) bf16 split ----------------
__global__ void split_bf16_kernel(const float* __restrict__ x,
                                  __bf16* __restrict__ hi,
                                  __bf16* __restrict__ lo, int n) {
    int i = blockIdx.x * blockDim.x + threadIdx.x;
    if (i < n) {
        float v = x[i];
        __bf16 h = (__bf16)v;
        hi[i] = h;
        lo[i] = (__bf16)(v - (float)h);
    }
}

// src is [Kdim][Ndim] f32; produce [Ndim][Kdim] hi/lo bf16 (row-major).
__global__ void split_bf16_T_kernel(const float* __restrict__ src,
                                    __bf16* __restrict__ hi,
                                    __bf16* __restrict__ lo,
                                    int Kdim, int Ndim) {
    int i = blockIdx.x * blockDim.x + threadIdx.x;   // index into [Ndim][Kdim]
    if (i < Kdim * Ndim) {
        int n = i / Kdim;
        int k = i - n * Kdim;
        float val = src[(size_t)k * Ndim + n];
        __bf16 h = (__bf16)val;
        hi[i] = h;
        lo[i] = (__bf16)(val - (float)h);
    }
}

// ---------------- fragment gather from LDS ----------------
__device__ __forceinline__ bf16x16 load_frag16(const __bf16* p0, const __bf16* p1) {
    bf16x8 a = *(const bf16x8*)p0;   // 16B aligned
    bf16x8 b = *(const bf16x8*)p1;   // 16B aligned
    return __builtin_shufflevector(a, b, 0,1,2,3,4,5,6,7,8,9,10,11,12,13,14,15);
}

// ---------------- generic split-bf16 GEMM:  C = epilogue(A @ B) ----------------
// A:   [M,K] f32 row-major (split to hi/lo bf16 on the fly through LDS)
// Bth/Btl: B pre-transposed, [N][K] bf16 row-major (pre-split hi/lo)
// MODE 0: C = acc + aux[n]            (aux = s,    len N)    -> s_mod
// MODE 1: C = acc * aux[m*N + n]      (aux = smod, [M,N])    -> low
// MODE 2: C = relu(acc + 2*aux[n])    (aux = bias, len N)    -> out
template <int MODE>
__global__ __launch_bounds__(256)
void gemm_bf16x2_kernel(const float* __restrict__ A,
                        const __bf16* __restrict__ Bth,
                        const __bf16* __restrict__ Btl,
                        float* __restrict__ C,
                        const float* __restrict__ aux,
                        int M, int N, int K) {
    constexpr int BM = 128, BN = 128, BK = 32;
    constexpr int LDK = BK + 8;              // pad rows to 80B: kills 4-way bank conflicts

    __shared__ __bf16 Ah[BM][LDK];
    __shared__ __bf16 Al[BM][LDK];
    __shared__ __bf16 Bhs[2][BN][LDK];       // ping-pong [buf][n][k]
    __shared__ __bf16 Bls[2][BN][LDK];

    const int tid  = threadIdx.x;
    const int lane = tid & 31;
    const int wid  = tid >> 5;
    const int wm   = wid & 3;                // 4 waves along M (32 rows each)
    const int wn   = wid >> 2;               // 2 waves along N (64 cols each)
    const int half = lane >> 4;
    const int l16  = lane & 15;

    f32x8 acc[2][4] = {};

    const size_t row0 = (size_t)blockIdx.y * BM;
    const size_t col0 = (size_t)blockIdx.x * BN;

    // A staging: 8 chunks of 4 f32 per 32-elem row; 4 chunks/thread
    const int r0 = tid >> 3;                 // base row, +32*i
    const int kc = (tid & 7) * 4;            // k offset of 4-element chunk
    // B staging: 16B chunks (8 bf16); 2 chunks/thread per matrix
    const int rB0 = tid >> 2;                // base row, +64*i
    const int c16 = (tid & 3) * 8;           // k offset of 8-element chunk

    const float*  Ap  = A   + row0 * K;
    const __bf16* Bhp = Bth + col0 * K;
    const __bf16* Blp = Btl + col0 * K;

    f32x4 a_reg[4];
#if !ASYNC_LDS
    u32x4 bh_reg[2], bl_reg[2];
#endif

    auto load_A = [&](int kb) {
        #pragma unroll
        for (int i = 0; i < 4; ++i) {
            size_t r = (size_t)(r0 + 32 * i);
            a_reg[i] = *(const f32x4*)(Ap + r * K + kb + kc);
        }
    };

#if ASYNC_LDS
    // gfx1250 async copy engine: global -> LDS, tracked by ASYNCcnt.
    auto async_B = [&](int kb, int buf) {
        #pragma unroll
        for (int i = 0; i < 2; ++i) {
            int r = rB0 + 64 * i;
            __builtin_amdgcn_global_load_async_to_lds_b128(
                (as1_v4i)(Bhp + (size_t)r * K + kb + c16),
                (as3_v4i)&Bhs[buf][r][c16], 0, 0);
            __builtin_amdgcn_global_load_async_to_lds_b128(
                (as1_v4i)(Blp + (size_t)r * K + kb + c16),
                (as3_v4i)&Bls[buf][r][c16], 0, 0);
        }
    };
#else
    auto load_B = [&](int kb) {
        #pragma unroll
        for (int i = 0; i < 2; ++i) {
            size_t r = (size_t)(rB0 + 64 * i);
            bh_reg[i] = *(const u32x4*)(Bhp + r * K + kb + c16);
            bl_reg[i] = *(const u32x4*)(Blp + r * K + kb + c16);
        }
    };
#endif

    load_A(0);
#if ASYNC_LDS
    async_B(0, 0);
#else
    load_B(0);
#endif

    int buf = 0;
    for (int kb = 0; kb < K; kb += BK) {
        // ---- stage A registers -> LDS (split f32 to hi/lo bf16) ----
        #pragma unroll
        for (int i = 0; i < 4; ++i) {
            int r = r0 + 32 * i;
            f32x4 x = a_reg[i];
            bf16x4 h, l;
            #pragma unroll
            for (int j = 0; j < 4; ++j) {
                h[j] = (__bf16)x[j];
                l[j] = (__bf16)(x[j] - (float)h[j]);
            }
            *(bf16x4*)&Ah[r][kc] = h;
            *(bf16x4*)&Al[r][kc] = l;
        }
#if ASYNC_LDS
        // this wave's async B(t) writes must land before the barrier publishes
#if __has_builtin(__builtin_amdgcn_s_wait_asynccnt)
        __builtin_amdgcn_s_wait_asynccnt(0);
#else
        asm volatile("s_wait_asynccnt 0x0" ::: "memory");
#endif
#else
        #pragma unroll
        for (int i = 0; i < 2; ++i) {
            int r = rB0 + 64 * i;
            *(u32x4*)&Bhs[buf][r][c16] = bh_reg[i];
            *(u32x4*)&Bls[buf][r][c16] = bl_reg[i];
        }
#endif
        __syncthreads();

        // ---- issue tile t+1 movement early (latency hidden by WMMA) ----
        if (kb + BK < K) {
            load_A(kb + BK);
#if ASYNC_LDS
            async_B(kb + BK, buf ^ 1);     // buffer last read at iter t-1: safe
#else
            load_B(kb + BK);
#endif
        }
        // ---- prefetch A tile t+2 into caches (global_prefetch_b8) ----
        if (kb + 2 * BK < K) {
            __builtin_prefetch(Ap + (size_t)r0 * K + kb + 2 * BK + kc, 0, 1);
        }

        // ---- gather fragments (wave32 WMMA VGPR layouts) ----
        bf16x16 afh[2], afl[2], bfh[4], bfl[4];
        #pragma unroll
        for (int mi = 0; mi < 2; ++mi) {
            int mr = wm * 32 + mi * 16 + l16;
            // A 16x32: lanes<16 hold K {0..7,16..23}; lanes>=16 hold {8..15,24..31}
            afh[mi] = load_frag16(&Ah[mr][half * 8], &Ah[mr][half * 8 + 16]);
            afl[mi] = load_frag16(&Al[mr][half * 8], &Al[mr][half * 8 + 16]);
        }
        #pragma unroll
        for (int ni = 0; ni < 4; ++ni) {
            int nc = wn * 64 + ni * 16 + l16;
            // B 32x16: lanes<16 hold K {0..7,16..23}; lanes>=16 hold {8..15,24..31}
            bfh[ni] = load_frag16(&Bhs[buf][nc][half * 8], &Bhs[buf][nc][half * 8 + 16]);
            bfl[ni] = load_frag16(&Bls[buf][nc][half * 8], &Bls[buf][nc][half * 8 + 16]);
        }

        // ---- split-precision MACs, 3 passes so consecutive wmma are independent
        #pragma unroll
        for (int mi = 0; mi < 2; ++mi)
            #pragma unroll
            for (int ni = 0; ni < 4; ++ni)
                acc[mi][ni] = __builtin_amdgcn_wmma_f32_16x16x32_bf16(
                    false, afh[mi], false, bfh[ni], (short)0, acc[mi][ni], false, false);
        #pragma unroll
        for (int mi = 0; mi < 2; ++mi)
            #pragma unroll
            for (int ni = 0; ni < 4; ++ni)
                acc[mi][ni] = __builtin_amdgcn_wmma_f32_16x16x32_bf16(
                    false, afl[mi], false, bfh[ni], (short)0, acc[mi][ni], false, false);
        #pragma unroll
        for (int mi = 0; mi < 2; ++mi)
            #pragma unroll
            for (int ni = 0; ni < 4; ++ni)
                acc[mi][ni] = __builtin_amdgcn_wmma_f32_16x16x32_bf16(
                    false, afh[mi], false, bfl[ni], (short)0, acc[mi][ni], false, false);

        __syncthreads();
        buf ^= 1;
    }

    // ---- epilogue + store (C/D layout: VGPR r -> M = half*8 + r, N = l16) ----
    #pragma unroll
    for (int mi = 0; mi < 2; ++mi) {
        #pragma unroll
        for (int ni = 0; ni < 4; ++ni) {
            #pragma unroll
            for (int r = 0; r < 8; ++r) {
                size_t m = row0 + wm * 32 + mi * 16 + half * 8 + r;
                size_t n = col0 + wn * 64 + ni * 16 + l16;
                float v = acc[mi][ni][r];
                if (MODE == 0) {
                    v += aux[n];
                } else if (MODE == 1) {
                    v *= aux[m * (size_t)N + n];
                } else {
                    v += 2.0f * aux[n];
                    v = v > 0.0f ? v : 0.0f;
                }
                C[m * (size_t)N + n] = v;
            }
        }
    }
}

// ---------------------------------------------------------------------------
extern "C" void kernel_launch(void* const* d_in, const int* in_sizes, int n_in,
                              void* d_out, int out_size, void* d_ws, size_t ws_size,
                              hipStream_t stream) {
    const float* data    = (const float*)d_in[0];   // [8192, 2048]
    const float* context = (const float*)d_in[1];   // [8192, 512]
    const float* u       = (const float*)d_in[2];   // [2048, 256]
    const float* s       = (const float*)d_in[3];   // [256]
    const float* v       = (const float*)d_in[4];   // [2048, 256]
    const float* w       = (const float*)d_in[5];   // [512, 256]
    const float* bias    = (const float*)d_in[6];   // [2048]
    float* out = (float*)d_out;                     // [8192, 2048]

    // ---- workspace layout ----
    char* ws = (char*)d_ws;
    size_t off = 0;
    float* smod = (float*)(ws + off); off += (size_t)BATCH * RANK_N * 4;       // 8 MB
    float* low  = (float*)(ws + off); off += (size_t)BATCH * RANK_N * 4;       // 8 MB
    // B factors pre-transposed to [N][K]:
    __bf16* ut_hi = (__bf16*)(ws + off); off += (size_t)RANK_N * NIN * 2;      // u^T [256][2048]
    __bf16* ut_lo = (__bf16*)(ws + off); off += (size_t)RANK_N * NIN * 2;
    __bf16* wt_hi = (__bf16*)(ws + off); off += (size_t)RANK_N * CCTX * 2;     // w^T [256][512]
    __bf16* wt_lo = (__bf16*)(ws + off); off += (size_t)RANK_N * CCTX * 2;
    __bf16* v_hi  = (__bf16*)(ws + off); off += (size_t)UNITS_N * RANK_N * 2;  // v = (v^T)^T [2048][256]
    __bf16* v_lo  = (__bf16*)(ws + off); off += (size_t)UNITS_N * RANK_N * 2;

    // ---- prep: split factors into hi/lo bf16 (B stored as [N][K]) ----
    {   // u [2048,256] -> u^T [256][2048]
        int n = NIN * RANK_N;
        split_bf16_T_kernel<<<(n + 255) / 256, 256, 0, stream>>>(u, ut_hi, ut_lo, NIN, RANK_N);
    }
    {   // w [512,256] -> w^T [256][512]
        int n = CCTX * RANK_N;
        split_bf16_T_kernel<<<(n + 255) / 256, 256, 0, stream>>>(w, wt_hi, wt_lo, CCTX, RANK_N);
    }
    {   // v [2048,256] is already (v^T) transposed -> straight split
        int n = UNITS_N * RANK_N;
        split_bf16_kernel<<<(n + 255) / 256, 256, 0, stream>>>(v, v_hi, v_lo, n);
    }

    // ---- GEMM 1: smod = context @ w + s        [8192,512]x[512,256] ----
    {
        dim3 grid(RANK_N / 128, BATCH / 128);
        gemm_bf16x2_kernel<0><<<grid, 256, 0, stream>>>(context, wt_hi, wt_lo, smod, s,
                                                        BATCH, RANK_N, CCTX);
    }
    // ---- GEMM 2: low = (data @ u) * smod       [8192,2048]x[2048,256] ----
    {
        dim3 grid(RANK_N / 128, BATCH / 128);
        gemm_bf16x2_kernel<1><<<grid, 256, 0, stream>>>(data, ut_hi, ut_lo, low, smod,
                                                        BATCH, RANK_N, NIN);
    }
    // ---- GEMM 3: out = relu(low @ v^T + 2b)    [8192,256]x[256,2048] ----
    {
        dim3 grid(UNITS_N / 128, BATCH / 128);
        gemm_bf16x2_kernel<2><<<grid, 256, 0, stream>>>(low, v_hi, v_lo, out, bias,
                                                        BATCH, UNITS_N, RANK_N);
    }
}
